// MLP_model_g_28166395527448
// MI455X (gfx1250) — matrix operand order repaired and checked by reading, not compiled, not run on hardware
//
#include <hip/hip_runtime.h>
#include <hip/hip_bf16.h>

#define N_ROWS     131072
#define D_IN_      512
#define D_H_       1024
#define D_OUT_     128
#define NUM_GROUPS 1024
#define EPS_       1e-5f

#define M_TILE   64    // rows per block (four 16-row WMMA tiles)
#define THREADS  256   // 8 waves of 32 (wave32)

typedef __attribute__((ext_vector_type(16))) __bf16       v16bf;
typedef __attribute__((ext_vector_type(8)))  float        v8f;
typedef __attribute__((ext_vector_type(4)))  unsigned int u32x4;

union ABFrag { u32x4 u[2]; v16bf v; };

__device__ __forceinline__ unsigned short f2bf(float f) {
    unsigned u = __float_as_uint(f);
    u += 0x7fffu + ((u >> 16) & 1u);          // round-to-nearest-even
    return (unsigned short)(u >> 16);
}
__device__ __forceinline__ float bf2f(unsigned short h) {
    return __uint_as_float(((unsigned)h) << 16);
}

// ---------------- weight transpose + fp32 -> bf16 ----------------
// W is [K][Nc] row-major fp32; Wt is [Nc][K] row-major bf16.
__global__ void cvtT_kernel(const float* __restrict__ W,
                            unsigned short* __restrict__ Wt,
                            int K, int Nc) {
    int total = K * Nc;
    for (int o = blockIdx.x * blockDim.x + threadIdx.x; o < total;
         o += gridDim.x * blockDim.x) {
        int k = o % K;
        int n = o / K;
        Wt[o] = f2bf(W[(size_t)k * Nc + n]);
    }
}

// ---------------- zero sums + counts ----------------
__global__ void zero_kernel(float* __restrict__ p, int n) {
    int i = blockIdx.x * blockDim.x + threadIdx.x;
    if (i < n) p[i] = 0.0f;
}

// ---------------- group counts histogram ----------------
__global__ void counts_kernel(const int* __restrict__ batch,
                              float* __restrict__ counts, int n) {
    int i = blockIdx.x * blockDim.x + threadIdx.x;
    if (i < n) atomicAdd(&counts[batch[i]], 1.0f);
}

// ---------------- fused 3-layer MLP + segment-sum ----------------
// WMMA fragment conventions (wave32, 16x16x32 bf16):
//   A (16x32): lanes 0-15 K=0..15, lanes 16-31 K=16..31,
//              row M = lane&15. Two 16B LDS loads per fragment.
//   B (32x16): Wt stored [N][K]; lanes 0-15 K=0..15, lanes 16-31 K=16..31,
//              col N = lane&15. One contiguous 32B global load per fragment,
//              SHARED across FOUR M-tiles (four accumulators per K pass).
//   C/D (16x16 f32): element r of v8f -> (M = r + 8*(lane>>4), N = lane&15).
__launch_bounds__(THREADS)
__global__ void mlp_fused_kernel(const float* __restrict__ X,
                                 const unsigned short* __restrict__ W1t,
                                 const float* __restrict__ b1,
                                 const float* __restrict__ g1,
                                 const float* __restrict__ be1,
                                 const unsigned short* __restrict__ W2t,
                                 const float* __restrict__ b2,
                                 const float* __restrict__ g2,
                                 const float* __restrict__ be2,
                                 const unsigned short* __restrict__ W3t,
                                 const float* __restrict__ b3,
                                 const int* __restrict__ batch,
                                 float* __restrict__ sums) {
    __shared__ unsigned short sA[M_TILE * D_H_];   // 128 KB
    __shared__ unsigned short sB[M_TILE * D_H_];   // 128 KB
    __shared__ float sRed[M_TILE][4][2];           // 2 KB
    __shared__ float sStat[M_TILE][2];             // mu, rsigma

    const int tid  = threadIdx.x;
    const int lane = tid & 31;
    const int wave = tid >> 5;       // 0..7
    const int half = lane >> 4;      // 0/1
    const int l16  = lane & 15;
    const int row0 = blockIdx.x * M_TILE;

    // ---- stage X tile (64 x 512 fp32 -> bf16, stride 512) into sA ----
    for (int i = tid; i < M_TILE * D_IN_; i += THREADS) {
        int r = i >> 9;          // /512
        int c = i & (D_IN_ - 1);
        sA[r * D_IN_ + c] = f2bf(X[(size_t)(row0 + r) * D_IN_ + c]);
    }
    __syncthreads();

// Quad-M K-loop: B fragment loaded once, used by four WMMAs (M tiles 0..3).
#define WMMA_K_LOOP4(A0, A1, A2, A3, SIN, STRIDE, WT, KDIM)                      \
    {                                                                            \
        const unsigned short* ap0_ = &SIN[(l16)      * (STRIDE)];                \
        const unsigned short* ap1_ = &SIN[(16 + l16) * (STRIDE)];                \
        const unsigned short* ap2_ = &SIN[(32 + l16) * (STRIDE)];                \
        const unsigned short* ap3_ = &SIN[(48 + l16) * (STRIDE)];                \
        const unsigned short* bp0 = &WT[(size_t)(n0 + l16) * (KDIM)];            \
        __builtin_prefetch(bp0 + 16 * (KDIM), 0, 1);  /* next n-tile row */      \
        _Pragma("unroll 4")                                                      \
        for (int kb = 0; kb < (KDIM); kb += 32) {                                \
            ABFrag a0, a1, a2, a3, b;                                            \
            const unsigned short* bp = bp0 + kb + half * 16;                     \
            b.u[0] = *(const u32x4*)(bp);                                        \
            b.u[1] = *(const u32x4*)(bp + 8);                                    \
            const unsigned short* p0 = ap0_ + kb + half * 16;                    \
            a0.u[0] = *(const u32x4*)(p0);                                       \
            a0.u[1] = *(const u32x4*)(p0 + 8);                                   \
            const unsigned short* p1 = ap1_ + kb + half * 16;                    \
            a1.u[0] = *(const u32x4*)(p1);                                       \
            a1.u[1] = *(const u32x4*)(p1 + 8);                                   \
            const unsigned short* p2 = ap2_ + kb + half * 16;                    \
            a2.u[0] = *(const u32x4*)(p2);                                       \
            a2.u[1] = *(const u32x4*)(p2 + 8);                                   \
            const unsigned short* p3 = ap3_ + kb + half * 16;                    \
            a3.u[0] = *(const u32x4*)(p3);                                       \
            a3.u[1] = *(const u32x4*)(p3 + 8);                                   \
            A0 = __builtin_amdgcn_wmma_f32_16x16x32_bf16(                        \
                false, a0.v, false, b.v, (short)0, A0, false, false);            \
            A1 = __builtin_amdgcn_wmma_f32_16x16x32_bf16(                        \
                false, a1.v, false, b.v, (short)0, A1, false, false);            \
            A2 = __builtin_amdgcn_wmma_f32_16x16x32_bf16(                        \
                false, a2.v, false, b.v, (short)0, A2, false, false);            \
            A3 = __builtin_amdgcn_wmma_f32_16x16x32_bf16(                        \
                false, a3.v, false, b.v, (short)0, A3, false, false);            \
        }                                                                        \
    }

// Bias + ReLU + bf16 store for one 16x16 tile whose top row is MBASE.
#define STORE_RELU_TILE(ACC, MBASE, DST, BIAS)                                  \
    {                                                                           \
        _Pragma("unroll")                                                       \
        for (int r = 0; r < 8; ++r) {                                           \
            float v = ACC[r] + (BIAS);                                          \
            v = v > 0.f ? v : 0.f;                                              \
            const int m = (MBASE) + r + half * 8;                               \
            DST[m * D_H_ + n0 + l16] = f2bf(v);                                 \
        }                                                                       \
    }

#define LAYERNORM(SRC, DST, GW, BW)                                             \
    {                                                                           \
        int r_ = tid >> 2;         /* 0..63 */                                  \
        int c0_ = tid & 3;                                                      \
        float s_ = 0.f, ss_ = 0.f;                                              \
        for (int c = c0_; c < D_H_; c += 4) {                                   \
            float x = bf2f(SRC[r_ * D_H_ + c]);                                 \
            s_ += x; ss_ += x * x;                                              \
        }                                                                       \
        sRed[r_][c0_][0] = s_;                                                  \
        sRed[r_][c0_][1] = ss_;                                                 \
        __syncthreads();                                                        \
        if (tid < M_TILE) {                                                     \
            float s2 = 0.f, ss2 = 0.f;                                          \
            _Pragma("unroll")                                                   \
            for (int i = 0; i < 4; ++i) { s2 += sRed[tid][i][0];                \
                                          ss2 += sRed[tid][i][1]; }             \
            float mu  = s2 * (1.0f / D_H_);                                     \
            float var = ss2 * (1.0f / D_H_) - mu * mu;                          \
            sStat[tid][0] = mu;                                                 \
            sStat[tid][1] = rsqrtf(var + EPS_);                                 \
        }                                                                       \
        __syncthreads();                                                        \
        for (int i = tid; i < M_TILE * D_H_; i += THREADS) {                    \
            int r = i >> 10;                                                    \
            int c = i & (D_H_ - 1);                                             \
            float x = bf2f(SRC[i]);                                             \
            float y = (x - sStat[r][0]) * sStat[r][1] * GW[c] + BW[c];          \
            DST[i] = f2bf(y);                                                   \
        }                                                                       \
        __syncthreads();                                                        \
    }

#define ZACC v8f{0.f, 0.f, 0.f, 0.f, 0.f, 0.f, 0.f, 0.f}

    // ================= layer 1: sB = relu(sA @ W1 + b1) =================
    for (int t = wave; t < D_H_ / 16; t += 8) {
        const int n0 = t * 16;
        v8f acc0 = ZACC, acc1 = ZACC, acc2 = ZACC, acc3 = ZACC;
        WMMA_K_LOOP4(acc0, acc1, acc2, acc3, sA, D_IN_, W1t, D_IN_)
        const float bias = b1[n0 + l16];
        STORE_RELU_TILE(acc0, 0,  sB, bias)
        STORE_RELU_TILE(acc1, 16, sB, bias)
        STORE_RELU_TILE(acc2, 32, sB, bias)
        STORE_RELU_TILE(acc3, 48, sB, bias)
    }
    __syncthreads();

    // ================= layernorm 1: sA = LN(sB; g1, be1) =================
    LAYERNORM(sB, sA, g1, be1)

    // ================= layer 2: sB = relu(sA @ W2 + b2) =================
    for (int t = wave; t < D_H_ / 16; t += 8) {
        const int n0 = t * 16;
        v8f acc0 = ZACC, acc1 = ZACC, acc2 = ZACC, acc3 = ZACC;
        WMMA_K_LOOP4(acc0, acc1, acc2, acc3, sA, D_H_, W2t, D_H_)
        const float bias = b2[n0 + l16];
        STORE_RELU_TILE(acc0, 0,  sB, bias)
        STORE_RELU_TILE(acc1, 16, sB, bias)
        STORE_RELU_TILE(acc2, 32, sB, bias)
        STORE_RELU_TILE(acc3, 48, sB, bias)
    }
    __syncthreads();

    // ================= layernorm 2: sA = LN(sB; g2, be2) =================
    LAYERNORM(sB, sA, g2, be2)

    // ================= layer 3 + segment-sum scatter =================
    {
        const int n0 = wave * 16;      // D_OUT/16 = 8 tiles, one per wave
        v8f acc0 = ZACC, acc1 = ZACC, acc2 = ZACC, acc3 = ZACC;
        WMMA_K_LOOP4(acc0, acc1, acc2, acc3, sA, D_H_, W3t, D_H_)
        const float bias = b3[n0 + l16];
#pragma unroll
        for (int r = 0; r < 8; ++r) {
            const int m0 = r + half * 8;
            {
                const int g = batch[row0 + m0];
                atomicAdd(&sums[(size_t)g * D_OUT_ + n0 + l16], acc0[r] + bias);
            }
            {
                const int g = batch[row0 + 16 + m0];
                atomicAdd(&sums[(size_t)g * D_OUT_ + n0 + l16], acc1[r] + bias);
            }
            {
                const int g = batch[row0 + 32 + m0];
                atomicAdd(&sums[(size_t)g * D_OUT_ + n0 + l16], acc2[r] + bias);
            }
            {
                const int g = batch[row0 + 48 + m0];
                atomicAdd(&sums[(size_t)g * D_OUT_ + n0 + l16], acc3[r] + bias);
            }
        }
    }
#undef WMMA_K_LOOP4
#undef STORE_RELU_TILE
#undef LAYERNORM
#undef ZACC
}

// ---------------- finalize: out = sums / counts ----------------
__global__ void finalize_kernel(const float* __restrict__ sums,
                                const float* __restrict__ counts,
                                float* __restrict__ out) {
    int i = blockIdx.x * blockDim.x + threadIdx.x;
    if (i < NUM_GROUPS * D_OUT_) {
        float c = counts[i / D_OUT_];
        out[i] = sums[i] / (c > 0.f ? c : 1.f);
    }
}

extern "C" void kernel_launch(void* const* d_in, const int* in_sizes, int n_in,
                              void* d_out, int out_size, void* d_ws, size_t ws_size,
                              hipStream_t stream) {
    const float* X   = (const float*)d_in[0];
    const float* W1  = (const float*)d_in[1];
    const float* b1  = (const float*)d_in[2];
    const float* g1  = (const float*)d_in[3];
    const float* be1 = (const float*)d_in[4];
    const float* W2  = (const float*)d_in[5];
    const float* b2  = (const float*)d_in[6];
    const float* g2  = (const float*)d_in[7];
    const float* be2 = (const float*)d_in[8];
    const float* W3  = (const float*)d_in[9];
    const float* b3  = (const float*)d_in[10];
    const int*   ab  = (const int*)d_in[11];
    float* out = (float*)d_out;

    // workspace layout (bytes)
    char* ws = (char*)d_ws;
    unsigned short* W1t = (unsigned short*)(ws);                            // 1,048,576 B
    unsigned short* W2t = (unsigned short*)(ws + 1048576);                  // 2,097,152 B
    unsigned short* W3t = (unsigned short*)(ws + 1048576 + 2097152);        //   262,144 B
    float* sums   = (float*)(ws + 1048576 + 2097152 + 262144);              //   524,288 B
    float* counts = sums + NUM_GROUPS * D_OUT_;                             //     4,096 B

    // 1) weight transpose/convert to bf16 [N][K]
    cvtT_kernel<<<1024, 256, 0, stream>>>(W1, W1t, D_IN_, D_H_);
    cvtT_kernel<<<1024, 256, 0, stream>>>(W2, W2t, D_H_,  D_H_);
    cvtT_kernel<<<512,  256, 0, stream>>>(W3, W3t, D_H_,  D_OUT_);

    // 2) zero accumulators (sums + counts are contiguous)
    const int nz = NUM_GROUPS * D_OUT_ + NUM_GROUPS;
    zero_kernel<<<(nz + 255) / 256, 256, 0, stream>>>(sums, nz);

    // 3) group counts
    counts_kernel<<<(N_ROWS + 255) / 256, 256, 0, stream>>>(ab, counts, N_ROWS);

    // 4) fused MLP + segment-sum
    mlp_fused_kernel<<<N_ROWS / M_TILE, THREADS, 0, stream>>>(
        X, W1t, b1, g1, be1, W2t, b2, g2, be2, W3t, b3, ab, sums);

    // 5) finalize segment mean
    finalize_kernel<<<(NUM_GROUPS * D_OUT_ + 255) / 256, 256, 0, stream>>>(
        sums, counts, out);
}
